// GNNEncoder_83700322664809
// MI455X (gfx1250) — matrix operand order repaired and checked by reading, not compiled, not run on hardware
//
#include <hip/hip_runtime.h>

// ---------------------------------------------------------------------------
// GNN encoder for MI455X (gfx1250): bf16 WMMA GEMMs + LDS-staged edge gather
// + f32 atomic scatter-add.  state kept in f32, gathered operand in bf16.
// Round 3: 64 edges/wave in the message GEMM (4 accumulator chains per
// double-buffered B fragment), register-resident dest ids, B double-buffer
// in all GEMMs.
// ---------------------------------------------------------------------------

typedef __bf16 bf16_t;
typedef __attribute__((ext_vector_type(16))) bf16_t      v16bf;
typedef __attribute__((ext_vector_type(8)))  float       v8f;
typedef __attribute__((ext_vector_type(4)))  unsigned int v4u;
typedef __attribute__((ext_vector_type(2)))  float       v2f;
typedef __attribute__((ext_vector_type(4)))  float       v4f;
typedef __attribute__((ext_vector_type(4)))  bf16_t      v4bf;

union FragU { v16bf v; v4u q[2]; };

// A fragment (16x32 bf16, MxK): lane L holds M=L&15; h=L>>4 selects K = 16h..16h+15.
__device__ __forceinline__ v16bf load_frag_a(const bf16_t* rowp, int k0, int h) {
    FragU f;
    f.q[0] = *(const v4u*)(rowp + k0 + h * 16);
    f.q[1] = *(const v4u*)(rowp + k0 + h * 16 + 8);
    return f.v;
}

// B fragment from pre-packed weights: 32 lanes x 16 bf16 contiguous per frag.
__device__ __forceinline__ v16bf load_frag_b(const bf16_t* wp, int frag, int lane) {
    const bf16_t* p = wp + ((size_t)frag * 32 + lane) * 16;
    FragU f;
    f.q[0] = *(const v4u*)(p);
    f.q[1] = *(const v4u*)(p + 8);
    return f.v;
}

__device__ __forceinline__ void wmma_bf16_step(v16bf a, v16bf b, v8f& acc) {
    acc = __builtin_amdgcn_wmma_f32_16x16x32_bf16(
        false, a, false, b, (short)0, acc, false, false);
}

// ---------------------------------------------------------------------------
// Pack a KxN row-major f32 weight matrix into bf16 WMMA-B fragments.
// packed[((n_tile*KT + k_tile)*32 + lane)*16 + e] = W[(k_tile*32+h*16+e)*N + n]
// ---------------------------------------------------------------------------
__global__ __launch_bounds__(256)
void pack_b_kernel(const float* __restrict__ W, bf16_t* __restrict__ Wp,
                   int K, int Ncols) {
    int t = blockIdx.x * blockDim.x + threadIdx.x;
    int total = K * Ncols;
    if (t >= total) return;
    int e      = t & 15;
    int lane   = (t >> 4) & 31;
    int frag   = t >> 9;
    int KT     = K >> 5;
    int k_tile = frag % KT;
    int n_tile = frag / KT;
    int h      = lane >> 4;
    int k      = k_tile * 32 + h * 16 + e;
    int n      = n_tile * 16 + (lane & 15);
    Wp[t] = (bf16_t)W[(size_t)k * Ncols + n];
}

// ---------------------------------------------------------------------------
// state = relu(x @ W_in + b_in);  writes f32 and bf16 copies.
// One wave per 16-node tile; 4 waves / block.  K=128 (4 k-tiles), N=256.
// ---------------------------------------------------------------------------
__global__ __launch_bounds__(128)
void input_gemm(const float* __restrict__ x, const bf16_t* __restrict__ Wp,
                const float* __restrict__ bias, float* __restrict__ state_f32,
                bf16_t* __restrict__ state_bf, int n_nodes) {
    __shared__ __align__(16) bf16_t sA[4][16][128];
    const int wave = threadIdx.x >> 5;
    const int lane = threadIdx.x & 31;
    const int tile = blockIdx.x * 4 + wave;
    const int ntiles = (n_nodes + 15) >> 4;
    if (tile >= ntiles) return;
    const int row0 = tile * 16;
    const bool full = (row0 + 16 <= n_nodes);

    bf16_t (*A)[128] = sA[wave];
    for (int t = lane; t < 16 * 64; t += 32) {
        int rr = t >> 6;
        int cc = (t & 63) * 2;
        int sr = row0 + rr; if (sr >= n_nodes) sr = n_nodes - 1;
        v2f xv = *(const v2f*)(x + (size_t)sr * 128 + cc);
        union { bf16_t b[2]; unsigned u; } pk;
        pk.b[0] = (bf16_t)xv.x;
        pk.b[1] = (bf16_t)xv.y;
        *(unsigned*)&A[rr][cc] = pk.u;
    }

    const int h = lane >> 4;
    const int m = lane & 15;
    const bf16_t* rowp = A[m];
    for (int ntile = 0; ntile < 16; ++ntile) {        // N = 256
        const int n = ntile * 16 + m;
        const float bv = bias[n];
        v8f acc = {bv, bv, bv, bv, bv, bv, bv, bv};
        v16bf b = load_frag_b(Wp, ntile * 4 + 0, lane);
#pragma unroll
        for (int kt = 0; kt < 4; ++kt) {              // K = 128
            v16bf bn = (kt < 3) ? load_frag_b(Wp, ntile * 4 + kt + 1, lane) : b;
            v16bf a = load_frag_a(rowp, kt * 32, h);
            wmma_bf16_step(a, b, acc);
            b = bn;
        }
        if (full) {
#pragma unroll
            for (int j = 0; j < 8; ++j) {
                int gr = row0 + h * 8 + j;
                float v = acc[j] > 0.f ? acc[j] : 0.f;
                state_f32[(size_t)gr * 256 + n] = v;
                state_bf[(size_t)gr * 256 + n]  = (bf16_t)v;
            }
        } else {
#pragma unroll
            for (int j = 0; j < 8; ++j) {
                int gr = row0 + h * 8 + j;
                if (gr < n_nodes) {
                    float v = acc[j] > 0.f ? acc[j] : 0.f;
                    state_f32[(size_t)gr * 256 + n] = v;
                    state_bf[(size_t)gr * 256 + n]  = (bf16_t)v;
                }
            }
        }
    }
}

// ---------------------------------------------------------------------------
// One message-passing round, deep M-blocking:
//   one wave handles 64 edges = four 16-row A tiles, four accumulator chains.
//   msg = relu(concat(state[src], state[dest]) @ W_msg[r] + b_msg[r])
//   state_f32[dest] += msg  (hardware f32 atomics)
// K=512 (16 k-tiles), N=256 (16 n-tiles).  Each double-buffered B fragment
// feeds 4 independent WMMAs; W_msg L2 traffic = 256KB per 64 edges.
// LDS = 64 KB exactly (gathered A tile); dest ids live in registers.
// ---------------------------------------------------------------------------
__global__ __launch_bounds__(32)
void msg_gemm(const int* __restrict__ edges, const bf16_t* __restrict__ state_bf,
              const bf16_t* __restrict__ Wp, const float* __restrict__ bias,
              float* __restrict__ state_f32, int n_edges) {
    __shared__ __align__(16) bf16_t sA[64][512];     // 64 KB
    const int lane = threadIdx.x & 31;
    const int e0 = blockIdx.x * 64;
    if (e0 >= n_edges) return;
    const bool full = (e0 + 64 <= n_edges);

    const int* srcIdx = edges;
    const int* dstIdx = edges + n_edges;

    // gather: row = edge-in-tile (0..63), cols 0..255 = state[src], 256..511 = state[dest]
    for (int t = lane; t < 64 * 64; t += 32) {       // 64 chunks of 8 bf16 per row
        int rr  = t >> 6;
        int col = (t & 63) * 8;
        int ee  = e0 + rr; if (ee >= n_edges) ee = n_edges - 1;
        int node, scol = col;
        if (col < 256) { node = srcIdx[ee]; }
        else           { node = dstIdx[ee]; scol = col - 256; }
        *(v4u*)&sA[rr][col] = *(const v4u*)(state_bf + (size_t)node * 256 + scol);
    }
    // wave-private LDS: in-wave ordering, no barrier needed

    const int h = lane >> 4;
    const int m = lane & 15;
    const bf16_t* rowp[4];
    int dn[4][8];
#pragma unroll
    for (int c = 0; c < 4; ++c) {
        rowp[c] = sA[c * 16 + m];
#pragma unroll
        for (int j = 0; j < 8; ++j) {
            int ee = e0 + c * 16 + h * 8 + j;
            if (ee >= n_edges) ee = n_edges - 1;
            dn[c][j] = dstIdx[ee];                   // L2-hot, redundant across lanes
        }
    }

    for (int ntile = 0; ntile < 16; ++ntile) {       // N = 256
        const int n = ntile * 16 + m;
        const float bv = bias[n];
        v8f acc[4];
#pragma unroll
        for (int c = 0; c < 4; ++c) {
            v8f z = {bv, bv, bv, bv, bv, bv, bv, bv};
            acc[c] = z;
        }
        v16bf b = load_frag_b(Wp, ntile * 16 + 0, lane);
#pragma unroll
        for (int kt = 0; kt < 16; ++kt) {            // K = 512
            v16bf bn = (kt < 15) ? load_frag_b(Wp, ntile * 16 + kt + 1, lane) : b;
#pragma unroll
            for (int c = 0; c < 4; ++c) {
                v16bf a = load_frag_a(rowp[c], kt * 32, h);
                wmma_bf16_step(a, b, acc[c]);
            }
            b = bn;
        }
        if (full) {
#pragma unroll
            for (int c = 0; c < 4; ++c)
#pragma unroll
                for (int j = 0; j < 8; ++j) {
                    float v = acc[c][j] > 0.f ? acc[c][j] : 0.f;
                    unsafeAtomicAdd(&state_f32[(size_t)dn[c][j] * 256 + n], v);
                }
        } else {
#pragma unroll
            for (int c = 0; c < 4; ++c)
#pragma unroll
                for (int j = 0; j < 8; ++j) {
                    if (e0 + c * 16 + h * 8 + j < n_edges) {
                        float v = acc[c][j] > 0.f ? acc[c][j] : 0.f;
                        unsafeAtomicAdd(&state_f32[(size_t)dn[c][j] * 256 + n], v);
                    }
                }
        }
    }
}

// state_bf = bf16(state_f32), vectorized x4
__global__ __launch_bounds__(256)
void cvt_kernel(const float* __restrict__ s, bf16_t* __restrict__ d, int n) {
    int t = (blockIdx.x * blockDim.x + threadIdx.x) * 4;
    if (t >= n) return;
    v4f v = *(const v4f*)(s + t);
    v4bf o = {(bf16_t)v.x, (bf16_t)v.y, (bf16_t)v.z, (bf16_t)v.w};
    *(v4bf*)(d + t) = o;
}

// ---------------------------------------------------------------------------
// out = state @ W_out + b_out.  K=256 (8 k-tiles), N=64 (4 n-tiles).
// ---------------------------------------------------------------------------
__global__ __launch_bounds__(128)
void out_gemm(const bf16_t* __restrict__ state_bf, const bf16_t* __restrict__ Wp,
              const float* __restrict__ bias, float* __restrict__ out, int n_nodes) {
    __shared__ __align__(16) bf16_t sA[4][16][256];
    const int wave = threadIdx.x >> 5;
    const int lane = threadIdx.x & 31;
    const int tile = blockIdx.x * 4 + wave;
    const int ntiles = (n_nodes + 15) >> 4;
    if (tile >= ntiles) return;
    const int row0 = tile * 16;
    const bool full = (row0 + 16 <= n_nodes);

    bf16_t (*A)[256] = sA[wave];
    for (int t = lane; t < 16 * 32; t += 32) {
        int rr = t >> 5;
        int cc = (t & 31) * 8;
        int sr = row0 + rr; if (sr >= n_nodes) sr = n_nodes - 1;
        *(v4u*)&A[rr][cc] = *(const v4u*)(state_bf + (size_t)sr * 256 + cc);
    }

    const int h = lane >> 4;
    const int m = lane & 15;
    const bf16_t* rowp = A[m];
    for (int ntile = 0; ntile < 4; ++ntile) {        // N = 64
        const int n = ntile * 16 + m;
        const float bv = bias[n];
        v8f acc = {bv, bv, bv, bv, bv, bv, bv, bv};
        v16bf b = load_frag_b(Wp, ntile * 8 + 0, lane);
#pragma unroll
        for (int kt = 0; kt < 8; ++kt) {             // K = 256
            v16bf bn = (kt < 7) ? load_frag_b(Wp, ntile * 8 + kt + 1, lane) : b;
            v16bf a = load_frag_a(rowp, kt * 32, h);
            wmma_bf16_step(a, b, acc);
            b = bn;
        }
        if (full) {
#pragma unroll
            for (int j = 0; j < 8; ++j)
                out[(size_t)(row0 + h * 8 + j) * 64 + n] = acc[j];
        } else {
#pragma unroll
            for (int j = 0; j < 8; ++j) {
                int gr = row0 + h * 8 + j;
                if (gr < n_nodes)
                    out[(size_t)gr * 64 + n] = acc[j];
            }
        }
    }
}

// ---------------------------------------------------------------------------
extern "C" void kernel_launch(void* const* d_in, const int* in_sizes, int n_in,
                              void* d_out, int out_size, void* d_ws, size_t ws_size,
                              hipStream_t stream) {
    const float* x     = (const float*)d_in[0];
    const int*   edges = (const int*)d_in[1];
    // d_in[2] = batch (unused by the reference computation)
    const float* W_in  = (const float*)d_in[3];
    const float* b_in  = (const float*)d_in[4];
    const float* W_msg = (const float*)d_in[5];   // [4, 512, 256]
    const float* b_msg = (const float*)d_in[6];   // [4, 256]
    const float* W_out = (const float*)d_in[7];
    const float* b_out = (const float*)d_in[8];
    float* out = (float*)d_out;

    const int F = 128, S = 256, L = 64, R = 4;
    const int N = in_sizes[0] / F;    // 25000 nodes
    const int E = in_sizes[1] / 2;    // 400000 edges
    (void)n_in; (void)out_size; (void)ws_size;

    // workspace carve-out (~40 MB)
    char* ws = (char*)d_ws;
    size_t off = 0;
    auto alignup = [](size_t v) { return (v + 255) & ~(size_t)255; };
    float*  state_f32 = (float*)(ws + off);  off = alignup(off + (size_t)N * S * 4);
    bf16_t* state_bf  = (bf16_t*)(ws + off); off = alignup(off + (size_t)N * S * 2);
    bf16_t* wp_in     = (bf16_t*)(ws + off); off = alignup(off + (size_t)F * S * 2);
    bf16_t* wp_msg    = (bf16_t*)(ws + off); off = alignup(off + (size_t)R * 2 * S * S * 2);
    bf16_t* wp_out    = (bf16_t*)(ws + off); off = alignup(off + (size_t)S * L * 2);

    // 1) pack weights into WMMA-B fragment layout (bf16)
    pack_b_kernel<<<(F * S + 255) / 256, 256, 0, stream>>>(W_in, wp_in, F, S);
    for (int r = 0; r < R; ++r)
        pack_b_kernel<<<(2 * S * S + 255) / 256, 256, 0, stream>>>(
            W_msg + (size_t)r * 2 * S * S, wp_msg + (size_t)r * 2 * S * S, 2 * S, S);
    pack_b_kernel<<<(S * L + 255) / 256, 256, 0, stream>>>(W_out, wp_out, S, L);

    // 2) input layer
    const int node_tiles = (N + 15) / 16;
    input_gemm<<<(node_tiles + 3) / 4, 128, 0, stream>>>(
        x, wp_in, b_in, state_f32, state_bf, N);

    // 3) message rounds
    const int edge_tiles64 = (E + 63) / 64;
    const int cvt_n = N * S;
    for (int r = 0; r < R; ++r) {
        msg_gemm<<<edge_tiles64, 32, 0, stream>>>(
            edges, state_bf, wp_msg + (size_t)r * 2 * S * S, b_msg + (size_t)r * S,
            state_f32, E);
        cvt_kernel<<<(cvt_n / 4 + 255) / 256, 256, 0, stream>>>(state_f32, state_bf, cvt_n);
    }

    // 4) output layer
    out_gemm<<<(node_tiles + 3) / 4, 128, 0, stream>>>(state_bf, wp_out, b_out, out, N);
}
